// PointNetSetAbstraction_3925600108913
// MI455X (gfx1250) — matrix operand order repaired and checked by reading, not compiled, not run on hardware
//
#include <hip/hip_runtime.h>
#include <hip/hip_bf16.h>

// ---------------------------------------------------------------------------
// PointNet Set Abstraction for MI455X (gfx1250, wave32, WMMA bf16)
// B=2, N=16384, NPOINT=4096, NSAMPLE=32, channels 67->64->64->128
//
// Layout strategy (all fragment loads become contiguous b128 vector loads):
//   activations: column-major [col][Kpad] bf16  -> B frag = 32 contiguous B
//   weights:     pre-packed in WMMA A-fragment order -> A frag = 32 contig B
//   GEMM output: column-major [col][Mout] bf16  -> D store = 16B uint4,
//                and it is already the next layer's B layout.
// ---------------------------------------------------------------------------

#define BATCH     2
#define NPTS      16384
#define NPOINT    4096
#define NSAMPLE   32
#define CIN       64
#define C0IN      67          // 3 + 64
#define C0PAD     96          // padded to multiple of 32 for wmma K
#define C1        64
#define C2        64
#define C3        128
#define COLS      (BATCH * NPOINT * NSAMPLE)   // 262144 global columns
#define EPSF      1e-05f
#define LEAKF     0.1f

typedef __attribute__((ext_vector_type(16))) __bf16 v16bf;
typedef __attribute__((ext_vector_type(8)))  float  v8f;

union Frag {
    v16bf v;
    uint4 q[2];
    unsigned short u[16];
};

// ---- bf16 <-> f32 helpers (bit-level, round-to-nearest-even) --------------
__device__ __forceinline__ unsigned short f2bf(float f) {
    unsigned int u = __float_as_uint(f);
    unsigned int r = u + 0x7FFFu + ((u >> 16) & 1u);
    return (unsigned short)(r >> 16);
}
__device__ __forceinline__ float bf2f(unsigned short b) {
    return __uint_as_float(((unsigned int)b) << 16);
}

// ===========================================================================
// zero stats scratch
// ===========================================================================
__global__ void zero_stats_kernel(float* s, int n) {
    int t = blockIdx.x * blockDim.x + threadIdx.x;
    if (t < n) s[t] = 0.0f;
}

// ===========================================================================
// brute-force KNN (top-32, unsorted -- downstream is order-invariant)
// ===========================================================================
#define KNN_T   128
#define KNN_TS  2048
__global__ void knn_kernel(const float* __restrict__ xyz, int* __restrict__ knn) {
    __shared__ float xs[KNN_TS], ys[KNN_TS], zs[KNN_TS];
    __shared__ float td[NSAMPLE * KNN_T];
    __shared__ int   ti[NSAMPLE * KNN_T];

    const int tid = threadIdx.x;
    const int q   = blockIdx.x * KNN_T + tid;
    const int b   = q >> 12;
    const int s   = q & (NPOINT - 1);
    const float* xb = xyz + (size_t)b * 3 * NPTS;

    const float qx = xb[0 * NPTS + s];
    const float qy = xb[1 * NPTS + s];
    const float qz = xb[2 * NPTS + s];

    for (int i = 0; i < NSAMPLE; ++i) {
        td[i * KNN_T + tid] = 3.4e38f;
        ti[i * KNN_T + tid] = 0;
    }
    float maxv = 3.4e38f;
    int   maxslot = 0;

    for (int n0 = 0; n0 < NPTS; n0 += KNN_TS) {
        for (int j = tid; j < KNN_TS; j += KNN_T) {
            xs[j] = xb[0 * NPTS + n0 + j];
            ys[j] = xb[1 * NPTS + n0 + j];
            zs[j] = xb[2 * NPTS + n0 + j];
        }
        __syncthreads();
        for (int j = 0; j < KNN_TS; ++j) {
            float dx = qx - xs[j], dy = qy - ys[j], dz = qz - zs[j];
            float d = dx * dx + dy * dy + dz * dz;
            if (d < maxv) {
                td[maxslot * KNN_T + tid] = d;
                ti[maxslot * KNN_T + tid] = n0 + j;
                float m = td[0 * KNN_T + tid]; int ms = 0;
                for (int i = 1; i < NSAMPLE; ++i) {
                    float t = td[i * KNN_T + tid];
                    if (t > m) { m = t; ms = i; }
                }
                maxv = m; maxslot = ms;
            }
        }
        __syncthreads();
    }
    for (int i = 0; i < NSAMPLE; ++i)
        knn[(size_t)q * NSAMPLE + i] = ti[i * KNN_T + tid];
}

// ===========================================================================
// copy new_xyz = xyz[:, :, :NPOINT] to output head
// ===========================================================================
__global__ void copy_newxyz_kernel(const float* __restrict__ xyz, float* __restrict__ out) {
    int t = blockIdx.x * blockDim.x + threadIdx.x;
    if (t >= BATCH * 3 * NPOINT) return;
    int b = t / (3 * NPOINT);
    int c = (t / NPOINT) % 3;
    int s = t % NPOINT;
    out[t] = xyz[(size_t)b * 3 * NPTS + (size_t)c * NPTS + s];
}

// ===========================================================================
// build x, column-major [col][C0PAD] bf16. One thread per column writes
// 96 contiguous bf16 (192 B). ch 0..2 direction, 3..66 points, rest zero.
// ===========================================================================
__global__ void build_x_kernel(const float* __restrict__ xyz,
                               const float* __restrict__ points,
                               const int* __restrict__ knn,
                               unsigned short* __restrict__ X) {
    int col = blockIdx.x * blockDim.x + threadIdx.x;
    if (col >= COLS) return;
    int s = (col >> 5) & (NPOINT - 1);
    int b = col >> 17;
    int idx = knn[col];

    unsigned short* xc = X + (size_t)col * C0PAD;
    const float* xb = xyz + (size_t)b * 3 * NPTS;
    #pragma unroll
    for (int c = 0; c < 3; ++c)
        xc[c] = f2bf(xb[c * NPTS + idx] - xb[c * NPTS + s]);

    const float* pb = points + (size_t)b * CIN * NPTS;
    #pragma unroll 8
    for (int c = 0; c < CIN; ++c)
        xc[3 + c] = f2bf(pb[(size_t)c * NPTS + idx]);

    #pragma unroll
    for (int c = C0IN; c < C0PAD; ++c)
        xc[c] = 0;
}

// ===========================================================================
// Pack weights into exact WMMA A-fragment order:
//   Apack[((tileM * KSTEPS + ks) * 32 + lane) * 16 + e]
// with K = half*16 + e (contiguous k order, same as the B fragment loads)
// M = lane & 15. Zero-padded past Kin. One thread per (tileM, ks, lane).
// ===========================================================================
__global__ void pack_w_kernel(const float* __restrict__ W,
                              unsigned short* __restrict__ Apack,
                              int Mout, int Kin, int KinPad) {
    int t = blockIdx.x * blockDim.x + threadIdx.x;
    int ksteps = KinPad >> 5;
    int total  = (Mout >> 4) * ksteps * 32;
    if (t >= total) return;
    int lane  = t & 31;
    int ks    = (t >> 5) % ksteps;
    int tileM = (t >> 5) / ksteps;
    int half  = lane >> 4;
    int row   = tileM * 16 + (lane & 15);
    unsigned short* dst = Apack + (size_t)t * 16;
    #pragma unroll
    for (int e = 0; e < 16; ++e) {
        int v  = e >> 1;
        int kb = half * 16 + e;
        int k  = ks * 32 + kb;
        dst[e] = (k < Kin) ? f2bf(W[(size_t)row * Kin + k]) : (unsigned short)0;
    }
}

// ===========================================================================
// WMMA GEMM:  Y[col][MOUT] = W (packed) * X[col][KPAD]
// One 16x16 output tile per wave; fully unrolled k loop; contiguous b128
// fragment loads; per-channel sum/sumsq via shfl reduction + f32 atomics.
// ===========================================================================
template <int KPAD, int MOUT>
__global__ void gemm_kernel(const unsigned short* __restrict__ Apack,
                            const unsigned short* __restrict__ X,
                            unsigned short* __restrict__ Y,
                            float* __restrict__ sum, float* __restrict__ sumsq) {
    constexpr int KSTEPS    = KPAD / 32;
    constexpr int NUMTILESM = MOUT / 16;

    const int lane   = threadIdx.x & 31;
    const int waveId = threadIdx.x >> 5;
    const int w      = blockIdx.x * 8 + waveId;
    const int tileM  = w % NUMTILESM;
    const int tileN  = w / NUMTILESM;
    const int n0     = tileN << 4;
    if (n0 >= COLS) return;
    const int m0   = tileM << 4;
    const int half = lane >> 4;
    const int nB   = n0 + (lane & 15);

    const unsigned short* acol = Apack + ((size_t)(tileM * KSTEPS) * 32 + lane) * 16;
    const unsigned short* bcol = X + (size_t)nB * KPAD + half * 16;

    v8f acc = {};
    #pragma unroll
    for (int ks = 0; ks < KSTEPS; ++ks) {
        Frag a, b;
        a.q[0] = *(const uint4*)(acol + (size_t)ks * 32 * 16);
        a.q[1] = *(const uint4*)(acol + (size_t)ks * 32 * 16 + 8);
        b.q[0] = *(const uint4*)(bcol + ks * 32);
        b.q[1] = *(const uint4*)(bcol + ks * 32 + 8);
        if (ks + 1 < KSTEPS)
            __builtin_prefetch(bcol + (ks + 1) * 32, 0, 3);
        acc = __builtin_amdgcn_wmma_f32_16x16x32_bf16(
            false, a.v, false, b.v, (short)0, acc, false, false);
    }

    // D tile store: lane's 8 rows are contiguous in column-major [col][MOUT]
    union { uint4 q; unsigned short u[8]; } outv;
    #pragma unroll
    for (int r = 0; r < 8; ++r)
        outv.u[r] = f2bf(acc[r]);
    *(uint4*)(Y + (size_t)nB * MOUT + m0 + 8 * half) = outv.q;

    // per-channel statistics (BN over all columns)
    #pragma unroll
    for (int r = 0; r < 8; ++r) {
        int m = m0 + r + 8 * half;
        float s = acc[r], s2 = acc[r] * acc[r];
        #pragma unroll
        for (int off = 1; off < 16; off <<= 1) {
            s  += __shfl_xor(s,  off, 32);
            s2 += __shfl_xor(s2, off, 32);
        }
        if ((lane & 15) == 0) {
            atomicAdd(&sum[m],   s);
            atomicAdd(&sumsq[m], s2);
        }
    }
}

// ===========================================================================
// per-channel BN coefficients  a = g*rsqrt(var+eps), c = b - mean*a
// ===========================================================================
__global__ void stats_kernel(const float* __restrict__ sum,
                             const float* __restrict__ sumsq,
                             const float* __restrict__ g,
                             const float* __restrict__ b,
                             float* __restrict__ a, float* __restrict__ c,
                             int C, float invCnt) {
    int ch = blockIdx.x * blockDim.x + threadIdx.x;
    if (ch >= C) return;
    float m = sum[ch] * invCnt;
    float v = sumsq[ch] * invCnt - m * m;
    float sc = g[ch] * rsqrtf(v + EPSF);
    a[ch] = sc;
    c[ch] = b[ch] - m * sc;
}

// ===========================================================================
// BN-apply + leaky-ReLU, elementwise in place on [col][64] bf16
// ===========================================================================
__global__ void bn_act_kernel(unsigned short* __restrict__ Y,
                              const float* __restrict__ a,
                              const float* __restrict__ c,
                              size_t total) {
    size_t t = (size_t)blockIdx.x * blockDim.x + threadIdx.x;
    if (t >= total) return;
    int m = (int)(t & 63);                 // MOUT == 64 for layers 0,1
    float v = a[m] * bf2f(Y[t]) + c[m];
    v = (v >= 0.0f) ? v : LEAKF * v;
    Y[t] = f2bf(v);
}

// ===========================================================================
// BN2 + leaky + max over NSAMPLE -> new_points   (Y2 layout [col][C3])
// ===========================================================================
__global__ void final_max_kernel(const unsigned short* __restrict__ Y2,
                                 const float* __restrict__ affA,
                                 const float* __restrict__ affC,
                                 float* __restrict__ out) {
    int t = blockIdx.x * blockDim.x + threadIdx.x;
    if (t >= BATCH * C3 * NPOINT) return;
    int b  = t / (C3 * NPOINT);
    int ch = (t / NPOINT) % C3;
    int s  = t % NPOINT;
    size_t base = ((size_t)(b * NPOINT + s) * NSAMPLE) * C3 + ch;
    float aa = affA[ch], cc = affC[ch];
    float m = -3.4e38f;
    #pragma unroll 8
    for (int k = 0; k < NSAMPLE; ++k) {
        float v = aa * bf2f(Y2[base + (size_t)k * C3]) + cc;
        v = (v >= 0.0f) ? v : LEAKF * v;
        m = fmaxf(m, v);
    }
    out[BATCH * 3 * NPOINT + t] = m;
}

// ===========================================================================
// Host launcher
// ===========================================================================
extern "C" void kernel_launch(void* const* d_in, const int* in_sizes, int n_in,
                              void* d_out, int out_size, void* d_ws, size_t ws_size,
                              hipStream_t stream) {
    const float* xyz    = (const float*)d_in[0];
    const float* points = (const float*)d_in[1];
    const float* W0     = (const float*)d_in[2];
    const float* W1     = (const float*)d_in[3];
    const float* W2     = (const float*)d_in[4];
    const float* g0 = (const float*)d_in[5];
    const float* b0 = (const float*)d_in[6];
    const float* g1 = (const float*)d_in[7];
    const float* b1 = (const float*)d_in[8];
    const float* g2 = (const float*)d_in[9];
    const float* b2 = (const float*)d_in[10];
    float* out = (float*)d_out;

    char* ws = (char*)d_ws;
    // workspace layout (all 16B aligned)
    size_t knnOff   = 0;
    size_t knnBytes = (size_t)COLS * sizeof(int);                         // 1.0 MB
    size_t statsOff = knnOff + knnBytes;
    float* S = (float*)(ws + statsOff);
    float *sum0 = S,        *sq0 = S + 128, *a0 = S + 256, *c0 = S + 384;
    float *sum1 = S + 512,  *sq1 = S + 640, *a1 = S + 768, *c1 = S + 896;
    float *sum2 = S + 1024, *sq2 = S + 1152,*a2 = S + 1280,*c2 = S + 1408;
    size_t xOff  = statsOff + 2048 * sizeof(float);
    size_t xB    = (size_t)COLS * C0PAD * sizeof(unsigned short);         // 50.3 MB
    size_t y0Off = xOff + xB;
    size_t yB    = (size_t)COLS * C1 * sizeof(unsigned short);            // 33.5 MB
    size_t y1Off = y0Off + yB;
    size_t y2Off = y1Off + yB;
    size_t y2B   = (size_t)COLS * C3 * sizeof(unsigned short);            // 67 MB
    size_t ap0Off = y2Off + y2B;
    size_t ap0B   = (size_t)(C1 / 16) * (C0PAD / 32) * 32 * 16 * 2;       // 12 KB
    size_t ap1Off = ap0Off + ((ap0B + 255) & ~(size_t)255);
    size_t ap1B   = (size_t)(C2 / 16) * (C1 / 32) * 32 * 16 * 2;
    size_t ap2Off = ap1Off + ((ap1B + 255) & ~(size_t)255);

    int* knn            = (int*)(ws + knnOff);
    unsigned short* X   = (unsigned short*)(ws + xOff);
    unsigned short* Y0  = (unsigned short*)(ws + y0Off);
    unsigned short* Y1  = (unsigned short*)(ws + y1Off);
    unsigned short* Y2  = (unsigned short*)(ws + y2Off);
    unsigned short* AP0 = (unsigned short*)(ws + ap0Off);
    unsigned short* AP1 = (unsigned short*)(ws + ap1Off);
    unsigned short* AP2 = (unsigned short*)(ws + ap2Off);

    const float invCnt = 1.0f / (float)COLS;

    // 0) zero stats, pack weights
    zero_stats_kernel<<<(1536 + 255) / 256, 256, 0, stream>>>(S, 1536);
    pack_w_kernel<<<2, 256, 0, stream>>>(W0, AP0, C1, C0IN, C0PAD);
    pack_w_kernel<<<1, 256, 0, stream>>>(W1, AP1, C2, C1, C1);
    pack_w_kernel<<<2, 256, 0, stream>>>(W2, AP2, C3, C2, C2);

    // 1) KNN
    knn_kernel<<<(BATCH * NPOINT) / KNN_T, KNN_T, 0, stream>>>(xyz, knn);

    // 2) new_xyz copy
    copy_newxyz_kernel<<<(BATCH * 3 * NPOINT + 255) / 256, 256, 0, stream>>>(xyz, out);

    // 3) gather + build x
    build_x_kernel<<<(COLS + 255) / 256, 256, 0, stream>>>(xyz, points, knn, X);

    // 4) layer0
    gemm_kernel<C0PAD, C1><<<((C1 / 16) * (COLS / 16)) / 8, 256, 0, stream>>>(
        AP0, X, Y0, sum0, sq0);
    stats_kernel<<<1, 128, 0, stream>>>(sum0, sq0, g0, b0, a0, c0, C1, invCnt);
    bn_act_kernel<<<(int)(((size_t)COLS * C1 + 255) / 256), 256, 0, stream>>>(
        Y0, a0, c0, (size_t)COLS * C1);

    // 5) layer1
    gemm_kernel<C1, C2><<<((C2 / 16) * (COLS / 16)) / 8, 256, 0, stream>>>(
        AP1, Y0, Y1, sum1, sq1);
    stats_kernel<<<1, 128, 0, stream>>>(sum1, sq1, g1, b1, a1, c1, C2, invCnt);
    bn_act_kernel<<<(int)(((size_t)COLS * C2 + 255) / 256), 256, 0, stream>>>(
        Y1, a1, c1, (size_t)COLS * C2);

    // 6) layer2
    gemm_kernel<C2, C3><<<((C3 / 16) * (COLS / 16)) / 8, 256, 0, stream>>>(
        AP2, Y1, Y2, sum2, sq2);
    stats_kernel<<<1, 128, 0, stream>>>(sum2, sq2, g2, b2, a2, c2, C3, invCnt);

    // 7) BN2 + leaky + max over samples
    final_max_kernel<<<(BATCH * C3 * NPOINT + 255) / 256, 256, 0, stream>>>(Y2, a2, c2, out);

    (void)in_sizes; (void)n_in; (void)out_size; (void)ws_size;
}
